// Mamba_31181462569227
// MI455X (gfx1250) — matrix operand order repaired and checked by reading, not compiled, not run on hardware
//
#include <hip/hip_runtime.h>

// ---------------------------------------------------------------------------
// Mamba forward for MI455X (gfx1250), wave32 + WMMA bf16.
// All GEMMs (in_proj, bcdelta, delta-proj, out_proj, tied logits) go through
// one v_wmma_f32_16x16x32_bf16 kernel. Sequential S6 scan done per (b,d,n)
// with 16-lane shuffle reduction. ~150MB of d_ws used.
// ---------------------------------------------------------------------------

typedef __bf16 bf16_t;
typedef __attribute__((ext_vector_type(16))) __bf16 v16bf;
typedef __attribute__((ext_vector_type(8)))  float  v8f;

#define NLAYERS 2
#define DMODEL  768
#define DINNER  1536
#define NSTATE  16
#define DRANK   48
#define KCONV   4
#define VOCAB   32000
#define BATCH   2
#define SEQ     1024
#define BL      (BATCH * SEQ)      // 2048
#define BCD_N   128                // 2N+DR = 80 padded to 128
#define DLT_K   64                 // DR = 48 padded to 64

// ---------------------------------------------------------------------------
// WMMA GEMM:  C[M x N] (f32) = A[M x K] (bf16 row-major, lda)
//                            x W[N x K]^T (bf16 row-major, ldw)
// grid = (N/128, M/64), block = 128 threads (4 waves: 2M x 2N wave grid).
// Each wave computes a 32x64 tile (2x4 of 16x16 WMMA tiles).
// Requires M%64==0, N%128==0, K%32==0, 16B-aligned rows.
// ---------------------------------------------------------------------------
__global__ __launch_bounds__(128)
void wmma_gemm_bf16(const bf16_t* __restrict__ A, int lda,
                    const bf16_t* __restrict__ W, int ldw,
                    float* __restrict__ C, int ldc, int K)
{
    const int lane = threadIdx.x & 31;
    const int wave = threadIdx.x >> 5;
    const int wm   = wave >> 1;            // 0..1
    const int wn   = wave & 1;             // 0..1
    const int mBase = blockIdx.y * 64 + wm * 32;
    const int nBase = blockIdx.x * 128 + wn * 64;
    const int lrow = lane & 15;
    const int kh   = lane >> 4;            // lane half (0/1)

    v8f acc[2][4];
#pragma unroll
    for (int i = 0; i < 2; ++i)
#pragma unroll
        for (int j = 0; j < 4; ++j) acc[i][j] = v8f{};

    const bf16_t* aRow[2];
    aRow[0] = A + (size_t)(mBase + lrow) * lda;
    aRow[1] = A + (size_t)(mBase + 16 + lrow) * lda;
    const bf16_t* wRow[4];
#pragma unroll
    for (int j = 0; j < 4; ++j)
        wRow[j] = W + (size_t)(nBase + j * 16 + lrow) * ldw;

    for (int k = 0; k < K; k += 32) {
        // prefetch next k-slab into caches (speculative, safe)
        __builtin_prefetch(aRow[0] + k + 32, 0, 0);
        __builtin_prefetch(wRow[0] + k + 32, 0, 0);
        __builtin_prefetch(wRow[2] + k + 32, 0, 0);

        v16bf a[2], b[4];
#pragma unroll
        for (int i = 0; i < 2; ++i) {
            // A frag: elems 0..7  -> K = k + kh*8 + (0..7)
            //         elems 8..15 -> K = k + 16 + kh*8 + (0..7)
            const bf16_t* p = aRow[i] + k + kh * 8;
            ((uint4*)&a[i])[0] = *(const uint4*)(p);
            ((uint4*)&a[i])[1] = *(const uint4*)(p + 16);
        }
#pragma unroll
        for (int j = 0; j < 4; ++j) {
            // B frag: elems 0..7 -> K = k + kh*8 + (0..7), elems 8..15 -> K = k + 16 + kh*8 + (0..7)
            const bf16_t* p = wRow[j] + k + kh * 8;
            ((uint4*)&b[j])[0] = *(const uint4*)(p);
            ((uint4*)&b[j])[1] = *(const uint4*)(p + 16);
        }
#pragma unroll
        for (int i = 0; i < 2; ++i)
#pragma unroll
            for (int j = 0; j < 4; ++j)
                acc[i][j] = __builtin_amdgcn_wmma_f32_16x16x32_bf16(
                    false, a[i], false, b[j], (short)0, acc[i][j], false, false);
    }

    // C/D layout: VGPR v, lane t -> M = v + 8*(t/16), N = t%16
#pragma unroll
    for (int i = 0; i < 2; ++i) {
        const int rb = mBase + i * 16 + 8 * kh;
#pragma unroll
        for (int j = 0; j < 4; ++j) {
            const int col = nBase + j * 16 + lrow;
            float* cp = C + (size_t)rb * ldc + col;
#pragma unroll
            for (int v = 0; v < 8; ++v) cp[(size_t)v * ldc] = acc[i][j][v];
        }
    }
}

// ---------------------------------------------------------------------------
// Elementwise / small kernels
// ---------------------------------------------------------------------------
__global__ __launch_bounds__(256)
void embed_kernel(const int* __restrict__ x, const float* __restrict__ emb,
                  float* __restrict__ h, int total)
{
    int i = blockIdx.x * 256 + threadIdx.x;
    if (i >= total) return;
    int row = i / DMODEL, c = i % DMODEL;
    h[i] = emb[(size_t)x[row] * DMODEL + c];
}

// fp32 -> bf16 with optional zero padding (rows/cols beyond input are zero)
__global__ __launch_bounds__(256)
void cvt_bf16_pad(const float* __restrict__ in, bf16_t* __restrict__ out,
                  int inRows, int inCols, int outCols, int total)
{
    int i = blockIdx.x * 256 + threadIdx.x;
    if (i >= total) return;
    int r = i / outCols, c = i % outCols;
    float v = (r < inRows && c < inCols) ? in[(size_t)r * inCols + c] : 0.f;
    out[i] = (bf16_t)v;
}

__global__ __launch_bounds__(256)
void rmsnorm_bf16_kernel(const float* __restrict__ h, const float* __restrict__ w,
                         bf16_t* __restrict__ out)
{
    __shared__ float red[256];
    const int r = blockIdx.x;
    const float* hp = h + (size_t)r * DMODEL;
    float s = 0.f;
    for (int c = threadIdx.x; c < DMODEL; c += 256) { float v = hp[c]; s += v * v; }
    red[threadIdx.x] = s;
    __syncthreads();
    for (int st = 128; st > 0; st >>= 1) {
        if (threadIdx.x < st) red[threadIdx.x] += red[threadIdx.x + st];
        __syncthreads();
    }
    const float scale = rsqrtf(red[0] / (float)DMODEL + 1e-5f);
    for (int c = threadIdx.x; c < DMODEL; c += 256)
        out[(size_t)r * DMODEL + c] = (bf16_t)(hp[c] * scale * w[c]);
}

// depthwise causal conv (K=4) + bias + silu; xi = first DINNER cols of xz
__global__ __launch_bounds__(256)
void conv_silu_kernel(const float* __restrict__ xz, const float* __restrict__ cw,
                      const float* __restrict__ cb, float* __restrict__ xc,
                      bf16_t* __restrict__ xcb, int total)
{
    int i = blockIdx.x * 256 + threadIdx.x;
    if (i >= total) return;
    const int d = i % DINNER;
    const int row = i / DINNER;        // b*SEQ + l
    const int l = row % SEQ;
    float acc = cb[d];
#pragma unroll
    for (int k = 0; k < KCONV; ++k) {
        int lp = l + k - (KCONV - 1);
        if (lp >= 0)
            acc += xz[(size_t)(row + k - (KCONV - 1)) * (2 * DINNER) + d] * cw[d * KCONV + k];
    }
    float s = acc / (1.f + __expf(-acc));      // silu
    xc[i]  = s;
    xcb[i] = (bf16_t)s;
}

// extract dlt = bcd[:, 32:80] -> bf16 (K padded to 64 with zeros)
__global__ __launch_bounds__(256)
void dlt_bf16_kernel(const float* __restrict__ bcd, bf16_t* __restrict__ out, int total)
{
    int i = blockIdx.x * 256 + threadIdx.x;
    if (i >= total) return;
    int r = i / DLT_K, c = i % DLT_K;
    float v = (c < DRANK) ? bcd[(size_t)r * BCD_N + 2 * NSTATE + c] : 0.f;
    out[i] = (bf16_t)v;
}

// delta = softplus(raw + dup_b[d])   (in place)
__global__ __launch_bounds__(256)
void softplus_kernel(float* __restrict__ delta, const float* __restrict__ db, int total)
{
    int i = blockIdx.x * 256 + threadIdx.x;
    if (i >= total) return;
    float v = delta[i] + db[i % DINNER];
    delta[i] = (v > 20.f) ? v : log1pf(__expf(v));
}

// sequential S6 scan: one thread per (b, d, n); 16-lane reduction over n
__global__ __launch_bounds__(256)
void scan_kernel(const float* __restrict__ delta, const float* __restrict__ bcd,
                 const float* __restrict__ xc, const float* __restrict__ lA,
                 float* __restrict__ ys)
{
    const int tid = threadIdx.x;
    const int n = tid & 15;
    const int d = blockIdx.x * 16 + (tid >> 4);
    const int b = blockIdx.y;
    const float An = -__expf(lA[d * NSTATE + n]);    // A = -exp(log(-A)) < 0
    float hs = 0.f;
    const size_t rowBase = (size_t)b * SEQ;
    for (int l = 0; l < SEQ; ++l) {
        const size_t row = rowBase + l;
        const float dv = delta[row * DINNER + d];
        const float Bn = bcd[row * BCD_N + n];
        const float Cn = bcd[row * BCD_N + NSTATE + n];
        const float xv = xc[row * DINNER + d];
        const float dA = dv * An;
        const float ab = __expf(dA);
        const float bb = ((fabsf(dA) > 1e-6f) ? (ab - 1.f) / dA : 1.f) * dv * Bn;
        hs = ab * hs + bb * xv;
        float y = Cn * hs;
        y += __shfl_xor(y, 1, 16);
        y += __shfl_xor(y, 2, 16);
        y += __shfl_xor(y, 4, 16);
        y += __shfl_xor(y, 8, 16);
        if (n == 0) ys[row * DINNER + d] = y;
    }
}

// g = (ys + xc * D[d]) * silu(res);  res = xz[:, DINNER + d]
__global__ __launch_bounds__(256)
void gate_kernel(const float* __restrict__ ys, const float* __restrict__ xc,
                 const float* __restrict__ Dp, const float* __restrict__ xz,
                 bf16_t* __restrict__ g, int total)
{
    int i = blockIdx.x * 256 + threadIdx.x;
    if (i >= total) return;
    const int d = i % DINNER;
    const int r = i / DINNER;
    const float res = xz[(size_t)r * (2 * DINNER) + DINNER + d];
    const float y = ys[i] + xc[i] * Dp[d];
    const float s = res / (1.f + __expf(-res));
    g[i] = (bf16_t)(y * s);
}

__global__ __launch_bounds__(256)
void add_kernel(float* __restrict__ h, const float* __restrict__ t, int total)
{
    int i = blockIdx.x * 256 + threadIdx.x;
    if (i < total) h[i] += t[i];
}

// ---------------------------------------------------------------------------
// Host orchestration
// ---------------------------------------------------------------------------
static inline dim3 egrid(int total) { return dim3((total + 255) / 256); }

extern "C" void kernel_launch(void* const* d_in, const int* in_sizes, int n_in,
                              void* d_out, int out_size, void* d_ws, size_t ws_size,
                              hipStream_t stream)
{
    const int*   x    = (const int*)  d_in[0];
    const float* emb  = (const float*)d_in[1];
    const float* rmsw = (const float*)d_in[2];
    const float* inw  = (const float*)d_in[3];
    const float* cw   = (const float*)d_in[4];
    const float* cb   = (const float*)d_in[5];
    const float* lA   = (const float*)d_in[6];
    const float* bcw  = (const float*)d_in[7];
    const float* dupw = (const float*)d_in[8];
    const float* dupb = (const float*)d_in[9];
    const float* Dp   = (const float*)d_in[10];
    const float* ow   = (const float*)d_in[11];
    const float* nfw  = (const float*)d_in[12];
    float* out = (float*)d_out;

    // bump allocator over workspace (~150 MB total)
    char* wsp = (char*)d_ws;
    auto alloc = [&](size_t bytes) -> void* {
        void* p = (void*)wsp;
        wsp += (bytes + 255) & ~(size_t)255;
        return p;
    };
    bf16_t* emb_bf  = (bf16_t*)alloc((size_t)VOCAB * DMODEL * 2);
    float*  h       = (float*) alloc((size_t)BL * DMODEL * 4);
    bf16_t* hn_bf   = (bf16_t*)alloc((size_t)BL * DMODEL * 2);
    bf16_t* wA_bf   = (bf16_t*)alloc((size_t)2 * DINNER * DMODEL * 2);
    float*  xz      = (float*) alloc((size_t)BL * 2 * DINNER * 4);
    float*  xc      = (float*) alloc((size_t)BL * DINNER * 4);
    bf16_t* xc_bf   = (bf16_t*)alloc((size_t)BL * DINNER * 2);
    bf16_t* bcw_bf  = (bf16_t*)alloc((size_t)BCD_N * DINNER * 2);
    float*  bcd     = (float*) alloc((size_t)BL * BCD_N * 4);
    bf16_t* dlt_bf  = (bf16_t*)alloc((size_t)BL * DLT_K * 2);
    bf16_t* dupw_bf = (bf16_t*)alloc((size_t)DINNER * DLT_K * 2);
    float*  delta   = (float*) alloc((size_t)BL * DINNER * 4);
    float*  ys      = (float*) alloc((size_t)BL * DINNER * 4);
    bf16_t* g_bf    = (bf16_t*)alloc((size_t)BL * DINNER * 2);
    bf16_t* ow_bf   = (bf16_t*)alloc((size_t)DMODEL * DINNER * 2);
    float*  tmp     = (float*) alloc((size_t)BL * DMODEL * 4);
    (void)ws_size; (void)n_in; (void)in_sizes; (void)out_size;

    // embedding -> bf16 once (49 MB, lives in L2 for the logits GEMM)
    {
        int total = VOCAB * DMODEL;
        cvt_bf16_pad<<<egrid(total), 256, 0, stream>>>(emb, emb_bf, VOCAB, DMODEL, DMODEL, total);
    }
    // token gather
    embed_kernel<<<egrid(BL * DMODEL), 256, 0, stream>>>(x, emb, h, BL * DMODEL);

    for (int layer = 0; layer < NLAYERS; ++layer) {
        const float* rw_l  = rmsw + (size_t)layer * DMODEL;
        const float* inw_l = inw  + (size_t)layer * 2 * DINNER * DMODEL;
        const float* cw_l  = cw   + (size_t)layer * DINNER * KCONV;
        const float* cb_l  = cb   + (size_t)layer * DINNER;
        const float* lA_l  = lA   + (size_t)layer * DINNER * NSTATE;
        const float* bcw_l = bcw  + (size_t)layer * (2 * NSTATE + DRANK) * DINNER;
        const float* dw_l  = dupw + (size_t)layer * DINNER * DRANK;
        const float* db_l  = dupb + (size_t)layer * DINNER;
        const float* Dp_l  = Dp   + (size_t)layer * DINNER;
        const float* ow_l  = ow   + (size_t)layer * DMODEL * DINNER;

        // rmsnorm -> bf16 activations
        rmsnorm_bf16_kernel<<<BL, 256, 0, stream>>>(h, rw_l, hn_bf);

        // in_proj: xz[2048 x 3072] = hn @ inw^T
        cvt_bf16_pad<<<egrid(2 * DINNER * DMODEL), 256, 0, stream>>>(
            inw_l, wA_bf, 2 * DINNER, DMODEL, DMODEL, 2 * DINNER * DMODEL);
        wmma_gemm_bf16<<<dim3((2 * DINNER) / 128, BL / 64), 128, 0, stream>>>(
            hn_bf, DMODEL, wA_bf, DMODEL, xz, 2 * DINNER, DMODEL);

        // causal conv + silu on xi
        conv_silu_kernel<<<egrid(BL * DINNER), 256, 0, stream>>>(
            xz, cw_l, cb_l, xc, xc_bf, BL * DINNER);

        // bcd[2048 x 128] = xc @ bcw^T  (N padded 80 -> 128)
        cvt_bf16_pad<<<egrid(BCD_N * DINNER), 256, 0, stream>>>(
            bcw_l, bcw_bf, 2 * NSTATE + DRANK, DINNER, DINNER, BCD_N * DINNER);
        wmma_gemm_bf16<<<dim3(BCD_N / 128, BL / 64), 128, 0, stream>>>(
            xc_bf, DINNER, bcw_bf, DINNER, bcd, BCD_N, DINNER);

        // delta_raw[2048 x 1536] = dlt @ dup_w^T  (K padded 48 -> 64)
        dlt_bf16_kernel<<<egrid(BL * DLT_K), 256, 0, stream>>>(bcd, dlt_bf, BL * DLT_K);
        cvt_bf16_pad<<<egrid(DINNER * DLT_K), 256, 0, stream>>>(
            dw_l, dupw_bf, DINNER, DRANK, DLT_K, DINNER * DLT_K);
        wmma_gemm_bf16<<<dim3(DINNER / 128, BL / 64), 128, 0, stream>>>(
            dlt_bf, DLT_K, dupw_bf, DLT_K, delta, DINNER, DLT_K);
        softplus_kernel<<<egrid(BL * DINNER), 256, 0, stream>>>(delta, db_l, BL * DINNER);

        // sequential selective scan
        scan_kernel<<<dim3(DINNER / 16, BATCH), 256, 0, stream>>>(delta, bcd, xc, lA_l, ys);

        // gating: g = (ys + xc*D) * silu(res)  -> bf16
        gate_kernel<<<egrid(BL * DINNER), 256, 0, stream>>>(ys, xc, Dp_l, xz, g_bf, BL * DINNER);

        // out_proj + residual
        cvt_bf16_pad<<<egrid(DMODEL * DINNER), 256, 0, stream>>>(
            ow_l, ow_bf, DMODEL, DINNER, DINNER, DMODEL * DINNER);
        wmma_gemm_bf16<<<dim3(DMODEL / 128, BL / 64), 128, 0, stream>>>(
            g_bf, DINNER, ow_bf, DINNER, tmp, DMODEL, DINNER);
        add_kernel<<<egrid(BL * DMODEL), 256, 0, stream>>>(h, tmp, BL * DMODEL);
    }

    // final norm + tied-embedding logits (the dominant GEMM: 2048x32000x768)
    rmsnorm_bf16_kernel<<<BL, 256, 0, stream>>>(h, nfw, hn_bf);
    wmma_gemm_bf16<<<dim3(VOCAB / 128, BL / 64), 128, 0, stream>>>(
        hn_bf, DMODEL, emb_bf, DMODEL, out, VOCAB, DMODEL);
}
